// GraphER_86878598463657
// MI455X (gfx1250) — matrix operand order repaired and checked by reading, not compiled, not run on hardware
//
#include <hip/hip_runtime.h>
#include <hip/hip_bf16.h>

typedef __bf16 bf16_t;
typedef __attribute__((ext_vector_type(16))) __bf16 v16bf;
typedef __attribute__((ext_vector_type(8)))  __bf16 v8bf;
typedef __attribute__((ext_vector_type(8)))  float  v8f;

#define N_NODES 50000
#define N_EDGES 1600000
#define N_CAND  131072
#define DH      128

// ---------------------------------------------------------------- utilities

__global__ void zero_f32_kernel(float* __restrict__ p, int n4) {
    int i = blockIdx.x * blockDim.x + threadIdx.x;
    if (i < n4) ((float4*)p)[i] = make_float4(0.f, 0.f, 0.f, 0.f);
}

__global__ void init_out_kernel(float* __restrict__ out, const float* __restrict__ b2) {
    int i = blockIdx.x * blockDim.x + threadIdx.x;
    if (i < N_CAND) out[i] = b2[0];
}

// ------------------------------------------------------ GIN neighborhood sum
// agg[dst] += x[src]; x resident in L2 (25.6MB), atomics resolve at L2.
__global__ void scatter_add_kernel(const float* __restrict__ x,
                                   const int* __restrict__ src,
                                   const int* __restrict__ dst,
                                   float* __restrict__ agg) {
    int t = blockIdx.x * blockDim.x + threadIdx.x;   // E * 32 threads
    int e = t >> 5;
    if (e >= N_EDGES) return;
    int c = (t & 31) << 2;
    int u = src[e], w = dst[e];
    float4 val = *(const float4*)(x + (size_t)u * DH + c);
    float* ap = agg + (size_t)w * DH + c;
    unsafeAtomicAdd(ap + 0, val.x);
    unsafeAtomicAdd(ap + 1, val.y);
    unsafeAtomicAdd(ap + 2, val.z);
    unsafeAtomicAdd(ap + 3, val.w);
}

// t_bf16 = (x + agg) converted to bf16 (GEMM A operand)
__global__ void add_convert_kernel(const float* __restrict__ x,
                                   const float* __restrict__ agg,
                                   bf16_t* __restrict__ tb, int n) {
    int i = blockIdx.x * blockDim.x + threadIdx.x;
    if (i < n) tb[i] = (bf16_t)(x[i] + agg[i]);
}

// -------------------------------------------- weight -> B-fragment swizzle
// B 32x16 bf16 layout: lane 0-15 -> n=lane, k=j (j<8) / 8+j (j>=8); lane 16-31 ->
// n=lane-16, k=8+j (j<8) / 16+j (j>=8). Fragment tile (kt, nt) stored as [32 lanes][16 bf16].
__global__ void prep_bfrag_kernel(const float* __restrict__ W,  // [K,N] row-major
                                  bf16_t* __restrict__ F, int K, int N) {
    int t = blockIdx.x * blockDim.x + threadIdx.x;
    if (t >= K * N) return;
    int j    = t & 15;
    int lane = (t >> 4) & 31;
    int tile = t >> 9;
    int ntiles = N >> 4;
    int nt = tile % ntiles;
    int kt = tile / ntiles;
    int n = nt * 16 + (lane & 15);
    int k = kt * 32 + ((lane < 16) ? 0 : 8) + ((j < 8) ? j : j + 8);
    F[t] = (bf16_t)W[(size_t)k * N + n];
}

// -------------------------------------------------------------- WMMA GEMM
// C[M,128] = act(A[M,K] @ B[K,128] + bias). One wave per 16x128 strip:
// A fragment loaded once per K-step, 8 WMMAs against 8 B fragments.
// A 16x32 bf16 fragment: lane row = lane%16; kbase = lane<16?0:8;
// elements 0..7 = A[row][kt*32+kbase .. +7], 8..15 = A[row][kt*32+kbase+16 ..].
__global__ void gemm_wmma_kernel(const bf16_t* __restrict__ A,
                                 const bf16_t* __restrict__ BF,
                                 const float* __restrict__ bias,
                                 float* __restrict__ Cf,     // optional f32 out
                                 bf16_t* __restrict__ Cb,    // optional bf16 out
                                 int M, int K, int relu) {
    const int lane  = threadIdx.x & 31;
    const int mtile = blockIdx.x * (blockDim.x >> 5) + (threadIdx.x >> 5);
    if ((mtile << 4) >= M) return;          // uniform per wave
    const int row   = (mtile << 4) + (lane & 15);
    const int kbase = (lane & 16) ? 8 : 0;

    const bf16_t* ap = A + (size_t)row * K + kbase;
    const bf16_t* bp = BF + (size_t)lane * 16;

    v8f acc[8] = {};
    const int ktiles = K >> 5;
    for (int kt = 0; kt < ktiles; ++kt) {
        v8bf alo = *(const v8bf*)(ap);
        v8bf ahi = *(const v8bf*)(ap + 16);
        v16bf a = __builtin_shufflevector(alo, ahi,
                     0,1,2,3,4,5,6,7,8,9,10,11,12,13,14,15);
#pragma unroll
        for (int nt = 0; nt < 8; ++nt) {
            v16bf b = *(const v16bf*)(bp + (size_t)nt * 512);
            acc[nt] = __builtin_amdgcn_wmma_f32_16x16x32_bf16(
                          false, a, false, b, (short)0, acc[nt], false, false);
        }
        ap += 32;
        bp += 8 * 512;                      // next kt block of fragments
    }

    const int rbase = (mtile << 4) + ((lane & 16) ? 8 : 0);
#pragma unroll
    for (int nt = 0; nt < 8; ++nt) {
        const int col  = (nt << 4) + (lane & 15);
        const float bv = bias[col];
#pragma unroll
        for (int r = 0; r < 8; ++r) {
            float v = acc[nt][r] + bv;
            if (relu) v = fmaxf(v, 0.0f);
            size_t off = (size_t)(rbase + r) * DH + col;
            if (Cf) Cf[off] = v;
            if (Cb) Cb[off] = (bf16_t)v;
        }
    }
}

// --------------------------------------------- edge predictor (fused WMMA)
// target part of feat folded into bias_eff; A fragments built on the fly from
// cand_rep = [x_u + x_v, |x_u - x_v|] (K=256), once per K-step and reused by
// all 8 N-tiles; epilogue fuses relu + dot(w2) with per-lane partial
// reduction over owned columns, then atomic adds into out (pre-init to b2).
__device__ __forceinline__ v8bf edge_rep8(const float* __restrict__ xu,
                                          const float* __restrict__ xv, int f) {
    v8bf r;
    if (f < DH) {
        v8f a = *(const v8f*)(xu + f);
        v8f b = *(const v8f*)(xv + f);
#pragma unroll
        for (int i = 0; i < 8; ++i) r[i] = (bf16_t)(a[i] + b[i]);
    } else {
        int d = f - DH;
        v8f a = *(const v8f*)(xu + d);
        v8f b = *(const v8f*)(xv + d);
#pragma unroll
        for (int i = 0; i < 8; ++i) r[i] = (bf16_t)__builtin_fabsf(a[i] - b[i]);
    }
    return r;
}

__global__ void edge_gemm_kernel(const float* __restrict__ x3,
                                 const int* __restrict__ cand,   // [C,2]
                                 const bf16_t* __restrict__ BF,  // K=256 frags
                                 const float* __restrict__ beff, // [128]
                                 const float* __restrict__ w2,   // [128]
                                 float* __restrict__ out) {      // [C]
    const int lane  = threadIdx.x & 31;
    const int mtile = blockIdx.x * (blockDim.x >> 5) + (threadIdx.x >> 5);
    const int row   = (mtile << 4) + (lane & 15);
    const int u = cand[2 * row];
    const int v = cand[2 * row + 1];
    const float* xu = x3 + (size_t)u * DH;
    const float* xv = x3 + (size_t)v * DH;
    const int kbase = (lane & 16) ? 8 : 0;
    const bf16_t* bp = BF + (size_t)lane * 16;

    v8f acc[8] = {};
#pragma unroll
    for (int kt = 0; kt < 8; ++kt) {        // K = 256
        int f0 = kt * 32 + kbase;
        v8bf alo = edge_rep8(xu, xv, f0);
        v8bf ahi = edge_rep8(xu, xv, f0 + 16);
        v16bf a = __builtin_shufflevector(alo, ahi,
                     0,1,2,3,4,5,6,7,8,9,10,11,12,13,14,15);
#pragma unroll
        for (int nt = 0; nt < 8; ++nt) {
            v16bf b = *(const v16bf*)(bp + (size_t)nt * 512);
            acc[nt] = __builtin_amdgcn_wmma_f32_16x16x32_bf16(
                          false, a, false, b, (short)0, acc[nt], false, false);
        }
        bp += 8 * 512;
    }

    // per-lane partial score over its 8 owned columns, then one atomic per row
    const int rbase = (mtile << 4) + ((lane & 16) ? 8 : 0);
    float part[8];
#pragma unroll
    for (int r = 0; r < 8; ++r) part[r] = 0.0f;
#pragma unroll
    for (int nt = 0; nt < 8; ++nt) {
        const int col  = (nt << 4) + (lane & 15);
        const float bv = beff[col];
        const float wv = w2[col];
#pragma unroll
        for (int r = 0; r < 8; ++r)
            part[r] += fmaxf(acc[nt][r] + bv, 0.0f) * wv;
    }
#pragma unroll
    for (int r = 0; r < 8; ++r)
        unsafeAtomicAdd(&out[rbase + r], part[r]);
}

// target_rep[256] = [x_u0 + x_v0, |x_u0 - x_v0|]
__global__ void target_rep_kernel(const float* __restrict__ x3,
                                  const int* __restrict__ pairs,
                                  float* __restrict__ tr) {
    int d = threadIdx.x;                    // 128 threads
    int u = pairs[0], v = pairs[1];
    float a = x3[(size_t)u * DH + d];
    float b = x3[(size_t)v * DH + d];
    tr[d]      = a + b;
    tr[DH + d] = fabsf(a - b);
}

// bias_eff[n] = b1[n] + sum_f target_rep[f] * pred_w1[f][n]  (f < 256)
__global__ void bias_eff_kernel(const float* __restrict__ tr,
                                const float* __restrict__ pw1, // [512,128]
                                const float* __restrict__ pb1,
                                float* __restrict__ beff) {
    int n = threadIdx.x;                    // 128 threads
    float s = pb1[n];
    for (int f = 0; f < 2 * DH; ++f) s += tr[f] * pw1[(size_t)f * DH + n];
    beff[n] = s;
}

// ------------------------------------------------------------------ driver

extern "C" void kernel_launch(void* const* d_in, const int* in_sizes, int n_in,
                              void* d_out, int out_size, void* d_ws, size_t ws_size,
                              hipStream_t stream) {
    const float* x    = (const float*)d_in[0];
    const int*   ei   = (const int*)d_in[1];    // [2, E]
    const int*   prs  = (const int*)d_in[2];    // [1, 2]
    const int*   cand = (const int*)d_in[3];    // [C, 2]
    const float* gw1  = (const float*)d_in[4];  // [3,128,128]
    const float* gb1  = (const float*)d_in[5];  // [3,128]
    const float* gw2  = (const float*)d_in[6];  // [3,128,128]
    const float* gb2  = (const float*)d_in[7];  // [3,128]
    const float* pw1  = (const float*)d_in[8];  // [512,128]
    const float* pb1  = (const float*)d_in[9];  // [128]
    const float* pw2  = (const float*)d_in[10]; // [128,1]
    const float* pb2  = (const float*)d_in[11]; // [1]
    float* out = (float*)d_out;

    // workspace carving (256B aligned)
    char* w = (char*)d_ws;
    size_t off = 0;
    auto carve = [&](size_t bytes) -> void* {
        void* p = w + off;
        off = (off + bytes + 255) & ~(size_t)255;
        return p;
    };
    float*  agg  = (float*)carve(sizeof(float) * (size_t)N_NODES * DH);
    float*  xcur = (float*)carve(sizeof(float) * (size_t)N_NODES * DH);
    bf16_t* tb   = (bf16_t*)carve(sizeof(bf16_t) * (size_t)N_NODES * DH);
    bf16_t* hb   = (bf16_t*)carve(sizeof(bf16_t) * (size_t)N_NODES * DH);
    bf16_t* w1f  = (bf16_t*)carve(sizeof(bf16_t) * 3 * DH * DH);
    bf16_t* w2f  = (bf16_t*)carve(sizeof(bf16_t) * 3 * DH * DH);
    bf16_t* pwf  = (bf16_t*)carve(sizeof(bf16_t) * 2 * DH * DH);
    float*  tr   = (float*)carve(sizeof(float) * 2 * DH);
    float*  beff = (float*)carve(sizeof(float) * DH);

    // pre-swizzle weights into WMMA B-fragment order
    for (int l = 0; l < 3; ++l) {
        prep_bfrag_kernel<<<(DH * DH + 255) / 256, 256, 0, stream>>>(
            gw1 + (size_t)l * DH * DH, w1f + (size_t)l * DH * DH, DH, DH);
        prep_bfrag_kernel<<<(DH * DH + 255) / 256, 256, 0, stream>>>(
            gw2 + (size_t)l * DH * DH, w2f + (size_t)l * DH * DH, DH, DH);
    }
    prep_bfrag_kernel<<<(2 * DH * DH + 255) / 256, 256, 0, stream>>>(
        pw1 + (size_t)2 * DH * DH, pwf, 2 * DH, DH);   // rows 256..511

    // GIN layers
    const float* xin = x;
    const int n_elem = N_NODES * DH;
    const int mtiles_node = N_NODES / 16;               // 3125
    const int gemm_blocks = (mtiles_node + 3) / 4;      // 4 waves / block
    for (int l = 0; l < 3; ++l) {
        zero_f32_kernel<<<(n_elem / 4 + 255) / 256, 256, 0, stream>>>(agg, n_elem / 4);
        scatter_add_kernel<<<(N_EDGES * 32) / 256, 256, 0, stream>>>(
            xin, ei, ei + N_EDGES, agg);
        add_convert_kernel<<<(n_elem + 255) / 256, 256, 0, stream>>>(
            xin, agg, tb, n_elem);
        // h = relu(t @ W1 + b1) -> bf16
        gemm_wmma_kernel<<<gemm_blocks, 128, 0, stream>>>(
            tb, w1f + (size_t)l * DH * DH, gb1 + (size_t)l * DH,
            nullptr, hb, N_NODES, DH, 1);
        // x' = h @ W2 + b2 -> f32
        gemm_wmma_kernel<<<gemm_blocks, 128, 0, stream>>>(
            hb, w2f + (size_t)l * DH * DH, gb2 + (size_t)l * DH,
            xcur, nullptr, N_NODES, DH, 0);
        xin = xcur;
    }

    // edge predictor
    target_rep_kernel<<<1, 128, 0, stream>>>(xcur, prs, tr);
    bias_eff_kernel<<<1, 128, 0, stream>>>(tr, pw1, pb1, beff);
    init_out_kernel<<<(N_CAND + 255) / 256, 256, 0, stream>>>(out, pb2);
    edge_gemm_kernel<<<(N_CAND / 16) / 4, 128, 0, stream>>>(
        xcur, cand, pwf, beff, pw2, out);
}
